// MultiHopCause_59399397704196
// MI455X (gfx1250) — matrix operand order repaired and checked by reading, not compiled, not run on hardware
//
#include <hip/hip_runtime.h>

// MI455X / gfx1250, wave32. BF16 WMMA (v_wmma_f32_16x16x32_bf16) with f32 accum.
// Block = 8 waves -> 128x64 output macro-tile; weight strip staged in LDS with a
// 520-element row skew (1040 B = 4 banks mod 64 -> conflict-free B-fragment reads).
// B fragments preloaded into distinct registers per K-step so ds_loads clause up
// and WMMAs issue back-to-back. Workspace requirement: ~197 MB (carved below).

typedef __attribute__((ext_vector_type(16))) __bf16 v16bf;
typedef __attribute__((ext_vector_type(8)))  float  v8f;

#define DD 512
#define BB 16
#define MM 1024
#define TT 4096
#define KT 1536            // 3*D for final GEMM
#define SLD 520            // LDS row stride (bf16 elems): 1040 B == 4 banks mod 64

#define WMMA_BF16(a, b, c) \
  __builtin_amdgcn_wmma_f32_16x16x32_bf16(false, (a), false, (b), (short)0, (c), false, false)

// ---------------- fragment loaders (ISA 7.12.2 16-bit layouts) ----------------

// A 16x32 bf16: lane<16 holds row (lane&15), K = k0 + {0..7, 16..23};
//               lane>=16 same rows,          K = k0 + {8..15, 24..31}.
static __device__ __forceinline__ v16bf ld_a_bf(const __bf16* arow, int k0, int lane) {
  const __bf16* p = arow + k0 + ((lane >> 4) << 3);
  union { uint4 u[2]; v16bf v; } f;
  f.u[0] = *(const uint4*)(p);        // 8 contiguous bf16 -> VGPR0..3
  f.u[1] = *(const uint4*)(p + 16);   // 8 contiguous bf16 -> VGPR4..7
  return f.v;
}

// Same A fragment sourced from f32 with a per-row scale, converted to bf16.
static __device__ __forceinline__ v16bf ld_a_f32s(const float* arow, int k0, int lane, float s) {
  const float* p = arow + k0 + ((lane >> 4) << 3);
  float4 x0 = *(const float4*)(p);
  float4 x1 = *(const float4*)(p + 4);
  float4 x2 = *(const float4*)(p + 16);
  float4 x3 = *(const float4*)(p + 20);
  v16bf v;
  v[0]  = (__bf16)(x0.x * s); v[1]  = (__bf16)(x0.y * s);
  v[2]  = (__bf16)(x0.z * s); v[3]  = (__bf16)(x0.w * s);
  v[4]  = (__bf16)(x1.x * s); v[5]  = (__bf16)(x1.y * s);
  v[6]  = (__bf16)(x1.z * s); v[7]  = (__bf16)(x1.w * s);
  v[8]  = (__bf16)(x2.x * s); v[9]  = (__bf16)(x2.y * s);
  v[10] = (__bf16)(x2.z * s); v[11] = (__bf16)(x2.w * s);
  v[12] = (__bf16)(x3.x * s); v[13] = (__bf16)(x3.y * s);
  v[14] = (__bf16)(x3.z * s); v[15] = (__bf16)(x3.w * s);
  return v;
}

// B 32x16 bf16 fragment from an LDS-staged weight strip S[64][SLD] (row = out col).
// lane = col n (lane&15); lanes<16 hold K = k0+{0..7,16..23}, lanes>=16 hold K = k0+{8..15,24..31}.
static __device__ __forceinline__ v16bf ld_b_lds(const __bf16* S, int c, int k0, int lane) {
  const __bf16* p = S + (size_t)c * SLD + k0 + ((lane >> 4) << 3);
  union { uint4 u[2]; v16bf v; } f;
  f.u[0] = *(const uint4*)(p);
  f.u[1] = *(const uint4*)(p + 16);
  return f.v;
}

// Stage 64 weight rows (= 64 output cols) x 512 K bf16 into LDS (skewed), 256 threads.
static __device__ __forceinline__ void stage_weights(const __bf16* __restrict__ W, int n0,
                                                     int kbase, int ldW, __bf16* S) {
  int tid = threadIdx.x;
  #pragma unroll
  for (int it = 0; it < 16; ++it) {
    int linear = it * 256 + tid;      // 4096 uint4 chunks total
    int row = linear >> 5;            // 32 uint4 per 512-bf16 row
    int c = (linear & 31) << 3;       // bf16 offset within row
    *(uint4*)(S + row * SLD + c) =
        *(const uint4*)(W + (size_t)(n0 + row) * ldW + kbase + c);
  }
}

// ---------------- elementwise / gather / scatter kernels ----------------

__global__ void cast_bf_kernel(const float* __restrict__ src, __bf16* __restrict__ dst, int n) {
  int i = blockIdx.x * blockDim.x + threadIdx.x;
  if (i < n) dst[i] = (__bf16)src[i];
}

__global__ void gather_cast_kernel(const float* __restrict__ emb, const int* __restrict__ ids,
                                   __bf16* __restrict__ out) {
  size_t idx = (size_t)blockIdx.x * blockDim.x + threadIdx.x;
  int d = (int)(idx & (DD - 1));
  size_t r = idx >> 9;
  int id = ids[r];
  out[idx] = (__bf16)emb[(size_t)id * DD + d];
}

__global__ void scatter_kernel(const __bf16* __restrict__ Xbf, const __bf16* __restrict__ Relbf,
                               const int* __restrict__ head, const int* __restrict__ tail,
                               const int* __restrict__ label, float* __restrict__ Upd) {
  size_t idx = (size_t)blockIdx.x * blockDim.x + threadIdx.x;   // over B*T*D
  int d = (int)(idx & (DD - 1));
  size_t bt = idx >> 9;
  if (label[bt] == -1) return;
  int b = (int)(bt >> 12);
  int h = head[bt], t = tail[bt];
  float relv = (float)Relbf[bt * DD + d];
  float xh = (float)Xbf[((size_t)(b * MM + h)) * DD + d];
  float xt = (float)Xbf[((size_t)(b * MM + t)) * DD + d];
  atomicAdd(Upd + ((size_t)(b * MM + t)) * DD + d, xh - relv);
  atomicAdd(Upd + ((size_t)(b * MM + h)) * DD + d, xt - relv);
}

__global__ void count_kernel(const int* __restrict__ head, const int* __restrict__ tail,
                             const int* __restrict__ label, float* __restrict__ Cnt) {
  int bt = blockIdx.x * blockDim.x + threadIdx.x;               // over B*T
  if (bt >= BB * TT) return;
  if (label[bt] == -1) return;
  int b = bt >> 12;
  atomicAdd(Cnt + b * MM + tail[bt], 1.0f);
  atomicAdd(Cnt + b * MM + head[bt], 1.0f);
}

__global__ void reduce_cause_kernel(const float* __restrict__ Trip, float* __restrict__ Cause) {
  int id = blockIdx.x * blockDim.x + threadIdx.x;               // over B*D
  int b = id >> 9, d = id & (DD - 1);
  const float* p = Trip + ((size_t)b * TT) * DD + d;
  float s = 0.0f;
  for (int t = 0; t < TT; ++t) s += p[(size_t)t * DD];
  Cause[id] = s;
}

// ---------------- WMMA GEMM kernels ----------------
// Geometry: grid.x = (rows/128)*(cols/64); block 256 threads = 8 waves.
// Wave w owns rows [bm*128 + w*16, +16) and all 64 cols (4 accumulators).

// Xout = bf16(relu(Xbf @ Ws^T + (Upd/max(cnt,1)) @ Wn^T)); rows = B*M, cols = D.
__global__ void gemm_node_kernel(const __bf16* __restrict__ Xbf, const float* __restrict__ Upd,
                                 const float* __restrict__ Cnt,
                                 const __bf16* __restrict__ Wsbf, const __bf16* __restrict__ Wnbf,
                                 __bf16* __restrict__ Xout) {
  extern __shared__ char smem_raw[];
  __bf16* S0 = (__bf16*)smem_raw;             // Ws strip 64xSLD
  __bf16* S1 = S0 + 64 * SLD;                 // Wn strip 64xSLD
  const int blocks_n = DD / 64;               // 8
  int bm = blockIdx.x / blocks_n, bn = blockIdx.x % blocks_n;
  int lane = threadIdx.x & 31, w = threadIdx.x >> 5;
  int n0 = bn * 64;

  stage_weights(Wsbf, n0, 0, DD, S0);
  stage_weights(Wnbf, n0, 0, DD, S1);
  __syncthreads();

  int row = bm * 128 + w * 16 + (lane & 15);
  const __bf16* arow = Xbf + (size_t)row * DD;
  const float*  urow = Upd + (size_t)row * DD;
  float s = 1.0f / fmaxf(Cnt[row], 1.0f);

  v8f acc[4];
  #pragma unroll
  for (int j = 0; j < 4; ++j) acc[j] = (v8f){0.f, 0.f, 0.f, 0.f, 0.f, 0.f, 0.f, 0.f};

  for (int k0 = 0; k0 < DD; k0 += 32) {
    v16bf a0 = ld_a_bf(arow, k0, lane);
    v16bf a1 = ld_a_f32s(urow, k0, lane, s);
    v16bf b0[4], b1[4];
    #pragma unroll
    for (int j = 0; j < 4; ++j) {
      int c = j * 16 + (lane & 15);
      b0[j] = ld_b_lds(S0, c, k0, lane);
      b1[j] = ld_b_lds(S1, c, k0, lane);
    }
    #pragma unroll
    for (int j = 0; j < 4; ++j) {
      acc[j] = WMMA_BF16(a0, b0[j], acc[j]);
      acc[j] = WMMA_BF16(a1, b1[j], acc[j]);
    }
  }

  int mb = bm * 128 + w * 16 + ((lane >> 4) << 3);
  #pragma unroll
  for (int j = 0; j < 4; ++j) {
    int n = n0 + j * 16 + (lane & 15);
    #pragma unroll
    for (int v = 0; v < 8; ++v) {
      float val = acc[j][v];
      Xout[(size_t)(mb + v) * DD + n] = (__bf16)(val > 0.0f ? val : 0.0f);
    }
  }
}

// Rout = bf16(Rin @ Wr^T); rows = B*T, cols = D.
__global__ void gemm_rel_kernel(const __bf16* __restrict__ Rin, const __bf16* __restrict__ Wrbf,
                                __bf16* __restrict__ Rout) {
  extern __shared__ char smem_raw[];
  __bf16* S = (__bf16*)smem_raw;              // 64xSLD
  const int blocks_n = DD / 64;
  int bm = blockIdx.x / blocks_n, bn = blockIdx.x % blocks_n;
  int lane = threadIdx.x & 31, w = threadIdx.x >> 5;
  int n0 = bn * 64;

  stage_weights(Wrbf, n0, 0, DD, S);
  __syncthreads();

  const __bf16* arow = Rin + (size_t)(bm * 128 + w * 16 + (lane & 15)) * DD;
  v8f acc[4];
  #pragma unroll
  for (int j = 0; j < 4; ++j) acc[j] = (v8f){0.f, 0.f, 0.f, 0.f, 0.f, 0.f, 0.f, 0.f};

  for (int k0 = 0; k0 < DD; k0 += 32) {
    v16bf a = ld_a_bf(arow, k0, lane);
    v16bf bfr[4];
    #pragma unroll
    for (int j = 0; j < 4; ++j)
      bfr[j] = ld_b_lds(S, j * 16 + (lane & 15), k0, lane);
    #pragma unroll
    for (int j = 0; j < 4; ++j)
      acc[j] = WMMA_BF16(a, bfr[j], acc[j]);
  }

  int mb = bm * 128 + w * 16 + ((lane >> 4) << 3);
  #pragma unroll
  for (int j = 0; j < 4; ++j) {
    int n = n0 + j * 16 + (lane & 15);
    #pragma unroll
    for (int v = 0; v < 8; ++v)
      Rout[(size_t)(mb + v) * DD + n] = (__bf16)acc[j][v];
  }
}

// Out[t,:] = concat(x[head[t]], rel[t], x[tail[t]]) @ Wt^T; K=1536 in 3 staged segments.
__global__ void gemm_trip_kernel(const __bf16* __restrict__ Xbf, const __bf16* __restrict__ Relbf,
                                 const __bf16* __restrict__ Wtbf,
                                 const int* __restrict__ head, const int* __restrict__ tail,
                                 float* __restrict__ Out) {
  extern __shared__ char smem_raw[];
  __bf16* S = (__bf16*)smem_raw;              // 64xSLD (one K segment)
  const int blocks_n = DD / 64;
  int bm = blockIdx.x / blocks_n, bn = blockIdx.x % blocks_n;
  int lane = threadIdx.x & 31, w = threadIdx.x >> 5;
  int n0 = bn * 64;

  int r = bm * 128 + w * 16 + (lane & 15);    // bt row
  int b = r >> 12;
  const __bf16* pH = Xbf + (size_t)(b * MM + head[r]) * DD;
  const __bf16* pR = Relbf + (size_t)r * DD;
  const __bf16* pT = Xbf + (size_t)(b * MM + tail[r]) * DD;

  v8f acc[4];
  #pragma unroll
  for (int j = 0; j < 4; ++j) acc[j] = (v8f){0.f, 0.f, 0.f, 0.f, 0.f, 0.f, 0.f, 0.f};

  for (int seg = 0; seg < 3; ++seg) {
    __syncthreads();                          // protect S from previous segment's readers
    stage_weights(Wtbf, n0, seg * DD, KT, S);
    __syncthreads();
    const __bf16* src = (seg == 0) ? pH : ((seg == 1) ? pR : pT);
    for (int k0 = 0; k0 < DD; k0 += 32) {
      v16bf a = ld_a_bf(src, k0, lane);
      v16bf bfr[4];
      #pragma unroll
      for (int j = 0; j < 4; ++j)
        bfr[j] = ld_b_lds(S, j * 16 + (lane & 15), k0, lane);
      #pragma unroll
      for (int j = 0; j < 4; ++j)
        acc[j] = WMMA_BF16(a, bfr[j], acc[j]);
    }
  }

  int mb = bm * 128 + w * 16 + ((lane >> 4) << 3);
  #pragma unroll
  for (int j = 0; j < 4; ++j) {
    int n = n0 + j * 16 + (lane & 15);
    #pragma unroll
    for (int v = 0; v < 8; ++v)
      Out[(size_t)(mb + v) * DD + n] = acc[j][v];
  }
}

// ---------------- host orchestration ----------------

extern "C" void kernel_launch(void* const* d_in, const int* in_sizes, int n_in,
                              void* d_out, int out_size, void* d_ws, size_t ws_size,
                              hipStream_t stream) {
  (void)in_sizes; (void)n_in; (void)out_size; (void)ws_size;
  const float* emb   = (const float*)d_in[0];   // [V, D]
  const float* Ws    = (const float*)d_in[1];   // [2, D, D]
  const float* Wn    = (const float*)d_in[2];   // [2, D, D]
  const float* Wr    = (const float*)d_in[3];   // [2, D, D]
  const float* Wt    = (const float*)d_in[4];   // [D, 3D]
  const int* cid     = (const int*)d_in[5];     // [B, M]
  const int* relid   = (const int*)d_in[6];     // [B, T]
  const int* head    = (const int*)d_in[7];     // [B, T]
  const int* tail    = (const int*)d_in[8];     // [B, T]
  const int* label   = (const int*)d_in[9];     // [B, T]
  float* out = (float*)d_out;                   // triple_repr [B*T*D] ++ cause [B*D]

  // workspace carve (256B aligned)
  char* p = (char*)d_ws;
  auto carve = [&](size_t bytes) { void* r = (void*)p; p += (bytes + 255) & ~(size_t)255; return r; };
  const size_t XN = (size_t)BB * MM * DD;       // 8.4M elems
  const size_t RN = (size_t)BB * TT * DD;       // 33.6M elems
  __bf16* xA   = (__bf16*)carve(XN * 2);
  __bf16* xB   = (__bf16*)carve(XN * 2);
  __bf16* relA = (__bf16*)carve(RN * 2);
  __bf16* relB = (__bf16*)carve(RN * 2);
  float*  upd  = (float*)carve(XN * 4);
  float*  cnt  = (float*)carve((size_t)BB * MM * 4);
  __bf16* wsb  = (__bf16*)carve((size_t)2 * DD * DD * 2);
  __bf16* wnb  = (__bf16*)carve((size_t)2 * DD * DD * 2);
  __bf16* wrb  = (__bf16*)carve((size_t)2 * DD * DD * 2);
  __bf16* wtb  = (__bf16*)carve((size_t)DD * KT * 2);

  const int TPB = 256;
  cast_bf_kernel<<<(2 * DD * DD + TPB - 1) / TPB, TPB, 0, stream>>>(Ws, wsb, 2 * DD * DD);
  cast_bf_kernel<<<(2 * DD * DD + TPB - 1) / TPB, TPB, 0, stream>>>(Wn, wnb, 2 * DD * DD);
  cast_bf_kernel<<<(2 * DD * DD + TPB - 1) / TPB, TPB, 0, stream>>>(Wr, wrb, 2 * DD * DD);
  cast_bf_kernel<<<(DD * KT + TPB - 1) / TPB, TPB, 0, stream>>>(Wt, wtb, DD * KT);
  gather_cast_kernel<<<(unsigned)(XN / TPB), TPB, 0, stream>>>(emb, cid, xA);
  gather_cast_kernel<<<(unsigned)(RN / TPB), TPB, 0, stream>>>(emb, relid, relA);

  __bf16* xin = xA;  __bf16* xout2 = xB;
  __bf16* rin = relA; __bf16* rout2 = relB;

  const int node_blocks = (BB * MM / 128) * (DD / 64);   // 128*8 = 1024
  const int rel_blocks  = (BB * TT / 128) * (DD / 64);   // 512*8 = 4096
  const size_t SM1 = (size_t)64 * SLD * sizeof(__bf16);  // 65 KB (skewed strip)
  const size_t SM2 = 2 * SM1;                            // 130 KB (node: Ws+Wn strips)

  for (int hop = 0; hop < 2; ++hop) {
    hipMemsetAsync(upd, 0, XN * 4, stream);
    hipMemsetAsync(cnt, 0, (size_t)BB * MM * 4, stream);
    scatter_kernel<<<(unsigned)(RN / TPB), TPB, 0, stream>>>(xin, rin, head, tail, label, upd);
    count_kernel<<<(BB * TT + TPB - 1) / TPB, TPB, 0, stream>>>(head, tail, label, cnt);
    gemm_node_kernel<<<node_blocks, TPB, SM2, stream>>>(
        xin, upd, cnt, wsb + (size_t)hop * DD * DD, wnb + (size_t)hop * DD * DD, xout2);
    gemm_rel_kernel<<<rel_blocks, TPB, SM1, stream>>>(
        rin, wrb + (size_t)hop * DD * DD, rout2);
    __bf16* t;
    t = xin; xin = xout2; xout2 = t;
    t = rin; rin = rout2; rout2 = t;
  }

  gemm_trip_kernel<<<rel_blocks, TPB, SM1, stream>>>(xin, rin, wtb, head, tail, out);
  reduce_cause_kernel<<<(BB * DD) / TPB, TPB, 0, stream>>>(out, out + (size_t)BB * TT * DD);
}
